// HGT2_52467320487958
// MI455X (gfx1250) — matrix operand order repaired and checked by reading, not compiled, not run on hardware
//
#include <hip/hip_runtime.h>
#include <hip/hip_bf16.h>
#include <math.h>

// ---------------------------------------------------------------------------
// HGT 2-layer forward for MI455X (gfx1250, wave32, WMMA).
//
//  * All dense linears (input lin, kqv, out) and the per-edge-type relation
//    transforms (hoisted to per-source-node GEMMs) run through one f16-WMMA
//    GEMM kernel (v_wmma_f32_16x16x32_f16, f32 accumulate).
//    - Block = 4 waves stacked in M; each wave owns a 16x64 C tile
//      (4 accumulators, A-fragment reuse). K is a template parameter so the
//      K loop fully unrolls (8/16 static WMMAs per kernel).
//    - B tile is cooperatively loaded (float4, coalesced), converted to f16
//      once, and stored TRANSPOSED in LDS (row stride K+8 halves: bank-
//      conflict-free, 16B aligned). Each B fragment = two ds_load_b128; all
//      four fragments of a K-chunk are loaded BEFORE the four WMMAs so DS
//      traffic overlaps matrix-op latency (descending s_wait_dscnt).
//    - A fragments: explicit float4 loads (global_load_b128) + packed cvt.
//  * Segment softmax over incoming edges: pass A computes per-edge logits and
//    an atomic max per (dst,head) via order-preserving uint encoding; pass B
//    computes e=exp(a-m), accumulates s, then scatter-adds v*e; the final
//    normalize divides by (s+1e-16) (identical to the reference's wgt).
//  * Workspace carve (~1.04 GB): h[2], kqv[2], one shared krel/vrel buffer,
//    agg[2], o[2], per-edge alpha, per-node m/s.
//
// Param flattening assumption (JAX tree order = sorted dict keys):
//   d_in[0..7]: x_paper, x_author, ei_cites_src/dst, ei_writes_src/dst,
//               ei_rev_src/dst
//   d_in[8..26]  = conv1: kqv.author.{b,w}, kqv.paper.{b,w},
//                  out.author.{b,w}, out.paper.{b,w},
//                  rel.cites.{a,m,p}, rel.rev_writes.{a,m,p},
//                  rel.writes.{a,m,p}, skip.author, skip.paper
//   d_in[27..45] = conv2 (same layout)
//   d_in[46..49] = lin.author.{b,w}, lin.paper.{b,w}
// ---------------------------------------------------------------------------

#define NP 200000
#define NA 100000
#define EC 500000
#define EW 300000
#define ER 300000

typedef __attribute__((ext_vector_type(16))) _Float16 v16h;
typedef __attribute__((ext_vector_type(8)))  _Float16 v8h;
typedef __attribute__((ext_vector_type(8)))  float    v8f;

// order-preserving float<->uint encoding for atomic max on signed floats
__device__ __forceinline__ unsigned encOrd(float f) {
  unsigned u = __float_as_uint(f);
  return (u & 0x80000000u) ? ~u : (u | 0x80000000u);
}
__device__ __forceinline__ float decOrd(unsigned x) {
  return (x & 0x80000000u) ? __uint_as_float(x & 0x7fffffffu)
                           : __uint_as_float(~x);
}
#define ENC_NEG_INF 0x007FFFFFu   // encOrd(-inf)

// ---------------------------------------------------------------------------
// GEMM: C[M,N](f32) = A[M,K](f32, lda, +aoff cols) * B[K,N](f32, ldb) + bias
// Requirements (hold at every call site): M%16==0, N%64==0, K in {64,128},
// all row strides / column offsets multiples of 4 floats (16B alignment).
// Grid: (ceil(M/64), N/64), block = 128 threads = 4 waves stacked in M.
// ---------------------------------------------------------------------------
template <int K>
__global__ void __launch_bounds__(128)
gemm_f16wmma(const float* __restrict__ A, int lda, int aoff,
             const float* __restrict__ B, int ldb,
             const float* __restrict__ bias,
             float* __restrict__ C, int ldc, int coff,
             int M, int N)
{
  constexpr int LDSB = K + 8;              // halves per LDS row (16B multiple)
  __shared__ _Float16 Bs[64 * LDSB];       // B tile, f16, TRANSPOSED [n][k]

  const int tid  = threadIdx.x;
  const int lane = tid & 31;
  const int wave = tid >> 5;
  const int m0 = (blockIdx.x * 4 + wave) * 16;
  const int n0 = blockIdx.y * 64;

  // ---- cooperative B tile load + f32->f16 + transpose into LDS ----
  // (all threads participate; coalesced float4 reads of B[k][n0..n0+63])
  for (int idx = tid; idx < K * 16; idx += 128) {
    const int k  = idx >> 4;
    const int c4 = (idx & 15) * 4;
    const float4 v = *(const float4*)(B + (size_t)k * ldb + n0 + c4);
    Bs[(c4 + 0) * LDSB + k] = (_Float16)v.x;
    Bs[(c4 + 1) * LDSB + k] = (_Float16)v.y;
    Bs[(c4 + 2) * LDSB + k] = (_Float16)v.z;
    Bs[(c4 + 3) * LDSB + k] = (_Float16)v.w;
  }
  __syncthreads();

  if (m0 < M) {
    const int r  = lane & 15;   // A row within tile / B+C column within tile
    const int hi = lane >> 4;   // lane half selects K / M sub-ranges
    const float* arow = A + (size_t)(m0 + r) * lda + aoff;
    v8f acc[4] = {{}, {}, {}, {}};
#pragma unroll
    for (int kb = 0; kb < K; kb += 32) {
      // A 16x32 f16 frag: lane r = row; half j -> K = kb+hi*16+j
      //                              (contiguous, matches B)
      const float4 a0 = *(const float4*)(arow + kb + hi * 16);
      const float4 a1 = *(const float4*)(arow + kb + hi * 16 + 4);
      const float4 a2 = *(const float4*)(arow + kb + hi * 16 + 8);
      const float4 a3 = *(const float4*)(arow + kb + hi * 16 + 12);
      v16h af;
      af[0]  = (_Float16)a0.x; af[1]  = (_Float16)a0.y;
      af[2]  = (_Float16)a0.z; af[3]  = (_Float16)a0.w;
      af[4]  = (_Float16)a1.x; af[5]  = (_Float16)a1.y;
      af[6]  = (_Float16)a1.z; af[7]  = (_Float16)a1.w;
      af[8]  = (_Float16)a2.x; af[9]  = (_Float16)a2.y;
      af[10] = (_Float16)a2.z; af[11] = (_Float16)a2.w;
      af[12] = (_Float16)a3.x; af[13] = (_Float16)a3.y;
      af[14] = (_Float16)a3.z; af[15] = (_Float16)a3.w;
      // Stage ALL four B fragments first (8x ds_load_b128), then issue the
      // four WMMAs: DS latency overlaps matrix-op latency.
      v16h bf[4];
#pragma unroll
      for (int t = 0; t < 4; ++t) {
        // B 32x16 f16 frag: lane r = column n0+t*16+r; half j -> K=kb+hi*16+j
        // -> 16 contiguous halves in transposed LDS: two ds_load_b128
        const _Float16* bp = &Bs[(t * 16 + r) * LDSB + kb + hi * 16];
        const v8h b0 = *(const v8h*)bp;
        const v8h b1 = *(const v8h*)(bp + 8);
        bf[t] = __builtin_shufflevector(b0, b1, 0, 1, 2, 3, 4, 5, 6, 7,
                                        8, 9, 10, 11, 12, 13, 14, 15);
      }
#pragma unroll
      for (int t = 0; t < 4; ++t)
        acc[t] = __builtin_amdgcn_wmma_f32_16x16x32_f16(false, af, false, bf[t],
                                                        (short)0, acc[t],
                                                        false, false);
    }
#pragma unroll
    for (int t = 0; t < 4; ++t) {
      const float bv = bias ? bias[n0 + t * 16 + r] : 0.0f;
#pragma unroll
      for (int rr = 0; rr < 8; ++rr) {      // C/D: m = rr + 8*hi, n = r
        const int m = m0 + rr + 8 * hi;
        C[(size_t)m * ldc + coff + n0 + t * 16 + r] = acc[t][rr] + bv;
      }
    }
  }
}

// ---------------------------------------------------------------------------
// Elementwise / edge kernels
// ---------------------------------------------------------------------------
__global__ void fill_u32_k(unsigned* p, unsigned v, int n) {
  int t = blockIdx.x * 256 + threadIdx.x;
  if (t < n) p[t] = v;
}
__global__ void fill_f32_k(float* p, float v, int n) {
  int t = blockIdx.x * 256 + threadIdx.x;
  if (t < n) p[t] = v;
}

// alpha[e,h] = (Q[dst,h,:] . krel[src,h,:]) * p[h] / sqrt(D);  atomic max m
__global__ void edge_pass_a_k(const int* __restrict__ src,
                              const int* __restrict__ dst, int E,
                              const float* __restrict__ kqvD,   // [Ndst,384]
                              const float* __restrict__ krel,   // [Nsrc,128]
                              const float* __restrict__ relp,   // [2]
                              float* __restrict__ alpha,        // [E,2]
                              unsigned* __restrict__ mEnc)      // [Ndst,2]
{
  int t = blockIdx.x * 256 + threadIdx.x;
  if (t >= E * 2) return;
  int e = t >> 1, h = t & 1;
  int s = src[e], d = dst[e];
  const float4* q4 = (const float4*)(kqvD + (size_t)d * 384 + 128 + h * 64);
  const float4* k4 = (const float4*)(krel + (size_t)s * 128 + h * 64);
  float acc = 0.f;
#pragma unroll
  for (int i = 0; i < 16; ++i) {
    float4 a = q4[i], b = k4[i];
    acc = fmaf(a.x, b.x, acc); acc = fmaf(a.y, b.y, acc);
    acc = fmaf(a.z, b.z, acc); acc = fmaf(a.w, b.w, acc);
  }
  float a = acc * relp[h] * 0.125f;                        // 1/sqrt(64)
  alpha[t] = a;
  atomicMax(&mEnc[(size_t)d * 2 + h], encOrd(a));
}

// decode m; nodes with no in-edges (-inf) -> 0  (matches reference where())
__global__ void finalize_m_k(unsigned* m, int n) {
  int t = blockIdx.x * 256 + threadIdx.x;
  if (t >= n) return;
  float f = decOrd(m[t]);
  if (!isfinite(f)) f = 0.0f;
  ((float*)m)[t] = f;
}

// alpha := exp(alpha - m[dst]);  s[dst] += alpha
__global__ void edge_pass_b_s_k(const int* __restrict__ dst, int E,
                                float* __restrict__ alpha,
                                const float* __restrict__ m,
                                float* __restrict__ s)
{
  int t = blockIdx.x * 256 + threadIdx.x;
  if (t >= E * 2) return;
  int e = t >> 1, h = t & 1;
  int d = dst[e];
  float ex = expf(alpha[t] - m[(size_t)d * 2 + h]);
  alpha[t] = ex;
  atomicAdd(&s[(size_t)d * 2 + h], ex);
}

// agg[dst,c] += vrel[src,c] * e   (c = h*64+d_idx, coalesced per edge)
__global__ void edge_pass_b_agg_k(const int* __restrict__ src,
                                  const int* __restrict__ dst, int E,
                                  const float* __restrict__ alpha,
                                  const float* __restrict__ vrel,
                                  float* __restrict__ agg)
{
  int t = blockIdx.x * 256 + threadIdx.x;
  if (t >= E * 128) return;
  int e = t >> 7, c = t & 127, h = c >> 6;
  int sN = src[e], d = dst[e];
  atomicAdd(&agg[(size_t)d * 128 + c],
            vrel[(size_t)sN * 128 + c] * alpha[(size_t)e * 2 + h]);
}

// agg := gelu(agg / (s + 1e-16))   (exact gelu, erf form)
__global__ void norm_gelu_k(float* __restrict__ agg,
                            const float* __restrict__ s, int total)
{
  int t = blockIdx.x * 256 + threadIdx.x;
  if (t >= total) return;
  int n = t >> 7, c = t & 127, h = c >> 6;
  float v = agg[t] / (s[(size_t)n * 2 + h] + 1e-16f);
  agg[t] = 0.5f * v * (1.0f + erff(v * 0.70710678118654752f));
}

// dst := sk*o + (1-sk)*h_old  [optional relu];  sk = sigmoid(skip scalar)
__global__ void skip_k(const float* __restrict__ o,
                       const float* __restrict__ hold,
                       const float* __restrict__ skp,
                       float* __restrict__ dst, int total, int relu)
{
  int t = blockIdx.x * 256 + threadIdx.x;
  if (t >= total) return;
  float sk = 1.0f / (1.0f + expf(-skp[0]));
  float r = sk * o[t] + (1.0f - sk) * hold[t];
  if (relu) r = fmaxf(r, 0.0f);
  dst[t] = r;
}

// ---------------------------------------------------------------------------
// Host orchestration
// ---------------------------------------------------------------------------
struct ConvP {
  const float *kqv_w_p, *kqv_b_p, *kqv_w_a, *kqv_b_a;
  const float *out_w_p, *out_b_p, *out_w_a, *out_b_a;
  const float *rel_a[3], *rel_m[3], *rel_p[3];   // index: 0=cites,1=writes,2=rev
  const float *skip_p, *skip_a;
};

static inline void gemmK128(hipStream_t st, const float* A, int lda, int aoff,
                            const float* B, int ldb, const float* bias,
                            float* C, int ldc, int coff, int M, int N) {
  dim3 g((M + 63) / 64, N / 64);
  gemm_f16wmma<128><<<g, 128, 0, st>>>(A, lda, aoff, B, ldb, bias, C, ldc, coff, M, N);
}
static inline void gemmK64(hipStream_t st, const float* A, int lda, int aoff,
                           const float* B, int ldb, const float* bias,
                           float* C, int ldc, int coff, int M, int N) {
  dim3 g((M + 63) / 64, N / 64);
  gemm_f16wmma<64><<<g, 128, 0, st>>>(A, lda, aoff, B, ldb, bias, C, ldc, coff, M, N);
}
static inline int nb(int n) { return (n + 255) / 256; }

static void run_conv(hipStream_t st, const float* hP, const float* hA,
                     const ConvP& cp,
                     float* kqv_p, float* kqv_a, float* relbuf,
                     float* agg_p, float* agg_a, float* o_p, float* o_a,
                     float* const al[3], const int* const esrc[3],
                     const int* const edst[3], const int eN[3],
                     const int eSrcT[3], const int eDstT[3],
                     unsigned* m_p, unsigned* m_a, float* s_p, float* s_a,
                     float* destP, float* destA, int relu)
{
  // kqv projections: [N,128] @ [128,384] + b
  gemmK128(st, hP, 128, 0, cp.kqv_w_p, 384, cp.kqv_b_p, kqv_p, 384, 0, NP, 384);
  gemmK128(st, hA, 128, 0, cp.kqv_w_a, 384, cp.kqv_b_a, kqv_a, 384, 0, NA, 384);

  fill_u32_k<<<nb(NP * 2), 256, 0, st>>>(m_p, ENC_NEG_INF, NP * 2);
  fill_u32_k<<<nb(NA * 2), 256, 0, st>>>(m_a, ENC_NEG_INF, NA * 2);
  fill_f32_k<<<nb(NP * 2), 256, 0, st>>>(s_p, 0.f, NP * 2);
  fill_f32_k<<<nb(NA * 2), 256, 0, st>>>(s_a, 0.f, NA * 2);
  fill_f32_k<<<nb(NP * 128), 256, 0, st>>>(agg_p, 0.f, NP * 128);
  fill_f32_k<<<nb(NA * 128), 256, 0, st>>>(agg_a, 0.f, NA * 128);

  const float* kqvT[2] = { kqv_p, kqv_a };
  unsigned*    mT[2]   = { m_p, m_a };
  float*       sT[2]   = { s_p, s_a };
  float*       aggT[2] = { agg_p, agg_a };
  const int    nT[2]   = { NP, NA };

  // Pass A: logits + segment max.  krel[src] = K[src] @ rel_a (per-head GEMM)
  for (int et = 0; et < 3; ++et) {
    int sT_ = eSrcT[et], dT_ = eDstT[et];
    for (int h = 0; h < 2; ++h)
      gemmK64(st, kqvT[sT_], 384, h * 64, cp.rel_a[et] + h * 64 * 64, 64,
              nullptr, relbuf, 128, h * 64, nT[sT_], 64);
    edge_pass_a_k<<<nb(eN[et] * 2), 256, 0, st>>>(
        esrc[et], edst[et], eN[et], kqvT[dT_], relbuf, cp.rel_p[et],
        al[et], mT[dT_]);
  }
  finalize_m_k<<<nb(NP * 2), 256, 0, st>>>(m_p, NP * 2);
  finalize_m_k<<<nb(NA * 2), 256, 0, st>>>(m_a, NA * 2);

  // Pass B1: e = exp(alpha - m), s += e
  for (int et = 0; et < 3; ++et)
    edge_pass_b_s_k<<<nb(eN[et] * 2), 256, 0, st>>>(
        edst[et], eN[et], al[et], (const float*)mT[eDstT[et]], sT[eDstT[et]]);

  // Pass B2: vrel[src] = V[src] @ rel_m, then agg[dst] += vrel[src]*e
  for (int et = 0; et < 3; ++et) {
    int sT_ = eSrcT[et], dT_ = eDstT[et];
    for (int h = 0; h < 2; ++h)
      gemmK64(st, kqvT[sT_], 384, 256 + h * 64, cp.rel_m[et] + h * 64 * 64, 64,
              nullptr, relbuf, 128, h * 64, nT[sT_], 64);
    edge_pass_b_agg_k<<<nb(eN[et] * 128), 256, 0, st>>>(
        esrc[et], edst[et], eN[et], al[et], relbuf, aggT[dT_]);
  }

  // normalize + gelu (in place), out linear, skip blend
  norm_gelu_k<<<nb(NP * 128), 256, 0, st>>>(agg_p, s_p, NP * 128);
  norm_gelu_k<<<nb(NA * 128), 256, 0, st>>>(agg_a, s_a, NA * 128);
  gemmK128(st, agg_p, 128, 0, cp.out_w_p, 128, cp.out_b_p, o_p, 128, 0, NP, 128);
  gemmK128(st, agg_a, 128, 0, cp.out_w_a, 128, cp.out_b_a, o_a, 128, 0, NA, 128);
  skip_k<<<nb(NP * 128), 256, 0, st>>>(o_p, hP, cp.skip_p, destP, NP * 128, relu);
  skip_k<<<nb(NA * 128), 256, 0, st>>>(o_a, hA, cp.skip_a, destA, NA * 128, relu);
}

extern "C" void kernel_launch(void* const* d_in, const int* in_sizes, int n_in,
                              void* d_out, int out_size, void* d_ws, size_t ws_size,
                              hipStream_t stream) {
  (void)in_sizes; (void)n_in; (void)out_size; (void)ws_size;
  const float* x_p = (const float*)d_in[0];
  const float* x_a = (const float*)d_in[1];
  const int* esrc[3] = { (const int*)d_in[2], (const int*)d_in[4], (const int*)d_in[6] };
  const int* edst[3] = { (const int*)d_in[3], (const int*)d_in[5], (const int*)d_in[7] };
  const int  eN[3]    = { EC, EW, ER };
  const int  eSrcT[3] = { 0, 1, 0 };   // cites: paper, writes: author, rev: paper
  const int  eDstT[3] = { 0, 0, 1 };   // cites: paper, writes: paper,  rev: author

  auto P = [&](int i) { return (const float*)d_in[i]; };
  ConvP c1, c2;
  auto fill = [&](ConvP& c, int b) {
    c.kqv_b_a = P(b + 0);  c.kqv_w_a = P(b + 1);
    c.kqv_b_p = P(b + 2);  c.kqv_w_p = P(b + 3);
    c.out_b_a = P(b + 4);  c.out_w_a = P(b + 5);
    c.out_b_p = P(b + 6);  c.out_w_p = P(b + 7);
    c.rel_a[0] = P(b + 8);  c.rel_m[0] = P(b + 9);  c.rel_p[0] = P(b + 10);  // cites
    c.rel_a[2] = P(b + 11); c.rel_m[2] = P(b + 12); c.rel_p[2] = P(b + 13);  // rev_writes
    c.rel_a[1] = P(b + 14); c.rel_m[1] = P(b + 15); c.rel_p[1] = P(b + 16);  // writes
    c.skip_a = P(b + 17);  c.skip_p = P(b + 18);
  };
  fill(c1, 8);
  fill(c2, 27);
  const float* lin_b_a = P(46); const float* lin_w_a = P(47);
  const float* lin_b_p = P(48); const float* lin_w_p = P(49);

  // workspace carve (~259.4M floats = ~1.04 GB)
  float* W = (float*)d_ws;
  size_t off = 0;
  auto take = [&](size_t n) { float* p = W + off; off += n; return p; };
  float* h_p    = take((size_t)NP * 128);
  float* h_a    = take((size_t)NA * 128);
  float* kqv_p  = take((size_t)NP * 384);
  float* kqv_a  = take((size_t)NA * 384);
  float* relbuf = take((size_t)NP * 128);   // shared krel/vrel staging
  float* agg_p  = take((size_t)NP * 128);
  float* agg_a  = take((size_t)NA * 128);
  float* o_p    = take((size_t)NP * 128);
  float* o_a    = take((size_t)NA * 128);
  float* al[3];
  al[0] = take((size_t)EC * 2);
  al[1] = take((size_t)EW * 2);
  al[2] = take((size_t)ER * 2);
  unsigned* m_p = (unsigned*)take((size_t)NP * 2);
  unsigned* m_a = (unsigned*)take((size_t)NA * 2);
  float* s_p    = take((size_t)NP * 2);
  float* s_a    = take((size_t)NA * 2);

  // input linears
  gemmK128(stream, x_p, 128, 0, lin_w_p, 128, lin_b_p, h_p, 128, 0, NP, 128);
  gemmK64 (stream, x_a,  64, 0, lin_w_a, 128, lin_b_a, h_a, 128, 0, NA, 128);

  // conv1 (writes back into h_p/h_a, relu) then conv2 (writes d_out, no relu)
  run_conv(stream, h_p, h_a, c1, kqv_p, kqv_a, relbuf, agg_p, agg_a, o_p, o_a,
           al, esrc, edst, eN, eSrcT, eDstT, m_p, m_a, s_p, s_a,
           h_p, h_a, 1);
  float* out_p = (float*)d_out;
  float* out_a = out_p + (size_t)NP * 128;
  run_conv(stream, h_p, h_a, c2, kqv_p, kqv_a, relbuf, agg_p, agg_a, o_p, o_a,
           al, esrc, edst, eN, eSrcT, eDstT, m_p, m_a, s_p, s_a,
           out_p, out_a, 0);
}
